// AttentionLSTM_5188320494083
// MI455X (gfx1250) — matrix operand order repaired and checked by reading, not compiled, not run on hardware
//
#include <hip/hip_runtime.h>
#include <cstdint>

typedef __bf16 bf16;
typedef __attribute__((ext_vector_type(16))) __bf16 v16bf;
typedef __attribute__((ext_vector_type(8)))  __bf16 v8bf;
typedef __attribute__((ext_vector_type(8)))  float  v8f;

#define NB   256
#define TT   128
#define DD   1024
#define HH   1024
#define LL   49
#define KTOT 3072            // D + H + H (concatenated GEMM K)
#define NOUT 4096            // 4*H
#define LDS_STRIDE 40        // 32 + 8 bf16 pad (80B rows, 16B aligned)

// ---- gfx1250 async global->LDS copy (ASYNCcnt path) -----------------------
__device__ __forceinline__ void async_ld_b128(unsigned lds_off, unsigned long long gaddr) {
  // per-lane: LDS[lds_off] = MEM[gaddr], 16 bytes
  asm volatile("global_load_async_to_lds_b128 %0, %1, off"
               :: "v"(lds_off), "v"(gaddr) : "memory");
}
__device__ __forceinline__ void wait_async0() {
  asm volatile("s_wait_asynccnt 0" ::: "memory");
}

// ---------------------------------------------------------------------------
// Once: stack + transpose weights to bf16, N-major: wT[j*3072 + k]
// k<1024 -> Wx[k][j]; k<2048 -> Wh[k-1024][j]; else Wattn[k-2048][j]
// ---------------------------------------------------------------------------
__global__ __launch_bounds__(256) void prep_weights_kernel(
    const float* __restrict__ Wx, const float* __restrict__ Wh,
    const float* __restrict__ Wa, bf16* __restrict__ wT) {
  size_t idx = (size_t)blockIdx.x * 256 + threadIdx.x;   // j*3072 + k
  int    k   = (int)(idx % KTOT);
  size_t j   = idx / KTOT;
  float v;
  if (k < DD)            v = Wx[(size_t)k * NOUT + j];
  else if (k < DD + HH)  v = Wh[(size_t)(k - DD) * NOUT + j];
  else                   v = Wa[(size_t)(k - DD - HH) * NOUT + j];
  wT[idx] = (bf16)v;
}

// ---------------------------------------------------------------------------
// Once: h0 = c0 = mean over 49 spatial elements of A
// ---------------------------------------------------------------------------
__global__ __launch_bounds__(256) void init_hc_kernel(
    const float* __restrict__ A, float* __restrict__ h, float* __restrict__ c) {
  size_t idx = (size_t)blockIdx.x * 256 + threadIdx.x;   // n*1024 + hh
  const float* Ap = A + idx * LL;
  float s = 0.f;
  #pragma unroll
  for (int l = 0; l < LL; ++l) s += Ap[l];
  s *= (1.0f / 49.0f);
  h[idx] = s;
  c[idx] = s;
}

// ---------------------------------------------------------------------------
// Per step: attention over L=49 + pack [xt | h | attn] as bf16 activations.
// One workgroup per batch row n.
// ---------------------------------------------------------------------------
__global__ __launch_bounds__(256) void attn_pack_kernel(
    const float* __restrict__ x, const float* __restrict__ A,
    const float* __restrict__ hprev, bf16* __restrict__ act, int t) {
  __shared__ float h_lds[HH];
  __shared__ float s_lds[LL];
  __shared__ float w_lds[LL];
  int n   = blockIdx.x;
  int tid = threadIdx.x;
  const float scale = 0.03125f;                      // 1/sqrt(1024)
  const float* An = A + (size_t)n * HH * LL;
  bf16* actn = act + (size_t)n * KTOT;

  // load h into LDS; pack x and h slices
  #pragma unroll
  for (int i = 0; i < 4; ++i) {
    int hh = tid * 4 + i;
    float hv = hprev[(size_t)n * HH + hh];
    h_lds[hh] = hv;
    actn[DD + hh] = (bf16)hv;
    float xv = x[((size_t)n * TT + t) * DD + hh];
    actn[hh] = (bf16)xv;
  }
  __syncthreads();

  // scores[l] = scale * sum_h A[n,h,l] * h[h]   (threads 0..48, coalesced in l)
  if (tid < LL) {
    float s = 0.f;
    for (int hh = 0; hh < HH; ++hh) s += An[(size_t)hh * LL + tid] * h_lds[hh];
    s_lds[tid] = s * scale;
  }
  __syncthreads();

  // softmax over 49 (redundant per-thread scan; trivial cost)
  if (tid < LL) {
    float m = -1e30f;
    for (int l = 0; l < LL; ++l) m = fmaxf(m, s_lds[l]);
    float sum = 0.f;
    for (int l = 0; l < LL; ++l) sum += __expf(s_lds[l] - m);
    w_lds[tid] = __expf(s_lds[tid] - m) / sum;
  }
  __syncthreads();

  // attn[h] = sum_l w[l]*A[n,h,l]; pack as bf16
  #pragma unroll
  for (int i = 0; i < 4; ++i) {
    int hh = tid * 4 + i;
    const float* Ah = An + (size_t)hh * LL;
    float a = 0.f;
    #pragma unroll 7
    for (int l = 0; l < LL; ++l) a += w_lds[l] * Ah[l];
    actn[DD + HH + hh] = (bf16)a;
  }
}

// ---------------------------------------------------------------------------
// Per step: a[256x4096] = act[256x3072] @ Wcat[3072x4096]  (bf16 WMMA, f32 acc)
// 64x64 macro tile, 8 waves, each wave = 16x32 strip (2 accumulators).
// Tiles staged by async global->LDS B128 copies, double-buffered, 1 barrier/it.
// ---------------------------------------------------------------------------
__global__ __launch_bounds__(256) void gemm_kernel(
    const bf16* __restrict__ act, const bf16* __restrict__ wT,
    float* __restrict__ abuf) {
  __shared__ __attribute__((aligned(16))) bf16 As[2][64 * LDS_STRIDE];
  __shared__ __attribute__((aligned(16))) bf16 Bs[2][64 * LDS_STRIDE];

  int tid  = threadIdx.x;
  int lane = tid & 31;
  int wave = tid >> 5;          // 0..7
  int wm   = wave >> 1;         // 0..3 -> 16-row strip within 64
  int wn   = wave & 1;          // 0..1 -> 32-col strip within 64
  int n0   = blockIdx.x * 64;   // 64 tiles over N=4096
  int m0   = blockIdx.y * 64;   // 4 tiles over M=256

  int lrow = tid >> 2;          // 0..63
  int lcol = (tid & 3) * 8;     // 0,8,16,24

  // per-lane source pointers (16B per lane per stage) and LDS destinations
  const bf16* aPtr = act + (size_t)(m0 + lrow) * KTOT + lcol;
  const bf16* bPtr = wT  + (size_t)(n0 + lrow) * KTOT + lcol;
  unsigned ldsA[2], ldsB[2];
  ldsA[0] = (unsigned)(uintptr_t)&As[0][lrow * LDS_STRIDE + lcol];
  ldsA[1] = (unsigned)(uintptr_t)&As[1][lrow * LDS_STRIDE + lcol];
  ldsB[0] = (unsigned)(uintptr_t)&Bs[0][lrow * LDS_STRIDE + lcol];
  ldsB[1] = (unsigned)(uintptr_t)&Bs[1][lrow * LDS_STRIDE + lcol];

  v8f acc0 = {};
  v8f acc1 = {};

  // prologue: fill stage 0
  async_ld_b128(ldsA[0], (unsigned long long)(uintptr_t)aPtr);
  async_ld_b128(ldsB[0], (unsigned long long)(uintptr_t)bPtr);

  const int NIT = KTOT / 32;    // 96
  for (int it = 0; it < NIT; ++it) {
    wait_async0();              // my wave's copies for stage it complete
    __syncthreads();            // all waves' copies visible; prev reads done
    if (it + 1 < NIT) {
      int nb = (it + 1) & 1;    // buffer (it+1)&1 was last read at stage it-1
      async_ld_b128(ldsA[nb], (unsigned long long)(uintptr_t)(aPtr + (size_t)(it + 1) * 32));
      async_ld_b128(ldsB[nb], (unsigned long long)(uintptr_t)(bPtr + (size_t)(it + 1) * 32));
      if (it + 2 < NIT)
        __builtin_prefetch(bPtr + (size_t)(it + 2) * 32, 0, 0);
    }

    const bf16* Asb = As[it & 1];
    const bf16* Bsb = Bs[it & 1];

    // A fragment (contiguous K, matching B): lane<16 -> K 0..15, lane>=16 -> K 16..31
    int mrow = wm * 16 + (lane & 15);
    int koff = (lane >> 4) * 16;
    union { v16bf v; v8bf h[2]; } fa, fb0, fb1;
    fa.h[0] = *(const v8bf*)&Asb[mrow * LDS_STRIDE + koff];
    fa.h[1] = *(const v8bf*)&Asb[mrow * LDS_STRIDE + koff + 8];

    // B fragments (32x16): lane&15 = column, lane>=16 -> K 16..31
    int bcol0 = wn * 32 + (lane & 15);
    int bcol1 = bcol0 + 16;
    int khalf = (lane >> 4) * 16;
    fb0.h[0] = *(const v8bf*)&Bsb[bcol0 * LDS_STRIDE + khalf];
    fb0.h[1] = *(const v8bf*)&Bsb[bcol0 * LDS_STRIDE + khalf + 8];
    fb1.h[0] = *(const v8bf*)&Bsb[bcol1 * LDS_STRIDE + khalf];
    fb1.h[1] = *(const v8bf*)&Bsb[bcol1 * LDS_STRIDE + khalf + 8];

    acc0 = __builtin_amdgcn_wmma_f32_16x16x32_bf16(
        false, fa.v, false, fb0.v, (short)0, acc0, false, false);
    acc1 = __builtin_amdgcn_wmma_f32_16x16x32_bf16(
        false, fa.v, false, fb1.v, (short)0, acc1, false, false);
  }

  // C layout: VGPR v, lane<16 -> M=v, lane>=16 -> M=v+8; N = lane&15
  int m     = m0 + wm * 16 + (lane >> 4) * 8;
  int nbase = n0 + wn * 32 + (lane & 15);
  #pragma unroll
  for (int v = 0; v < 8; ++v) {
    abuf[(size_t)(m + v) * NOUT + nbase]      = acc0[v];
    abuf[(size_t)(m + v) * NOUT + nbase + 16] = acc1[v];
  }
}

// ---------------------------------------------------------------------------
// Per step: gates + cell/hidden update + output write
// ---------------------------------------------------------------------------
__global__ __launch_bounds__(256) void gates_kernel(
    const float* __restrict__ abuf, const float* __restrict__ b,
    float* __restrict__ h, float* __restrict__ c,
    float* __restrict__ out, int t) {
  int idx = blockIdx.x * 256 + threadIdx.x;   // n*1024 + hh
  int n   = idx >> 10;
  int hh  = idx & 1023;
  const float* an = abuf + (size_t)n * NOUT;
  float ai = an[hh]            + b[hh];
  float af = an[HH + hh]       + b[HH + hh];
  float ao = an[2 * HH + hh]   + b[2 * HH + hh];
  float ag = an[3 * HH + hh]   + b[3 * HH + hh];
  float i_ = 1.f / (1.f + __expf(-ai));
  float f_ = 1.f / (1.f + __expf(-af));
  float o_ = 1.f / (1.f + __expf(-ao));
  float g_ = tanhf(ag);
  float cn = f_ * c[idx] + i_ * g_;
  float hn = o_ * tanhf(cn);
  c[idx] = cn;
  h[idx] = hn;
  out[((size_t)n * TT + t) * HH + hh] = hn;
}

// ---------------------------------------------------------------------------
extern "C" void kernel_launch(void* const* d_in, const int* in_sizes, int n_in,
                              void* d_out, int out_size, void* d_ws, size_t ws_size,
                              hipStream_t stream) {
  const float* x  = (const float*)d_in[0];   // (256,128,1024)
  const float* A  = (const float*)d_in[1];   // (256,1024,7,7)
  const float* Wx = (const float*)d_in[2];   // (1024,4096)
  const float* Wh = (const float*)d_in[3];   // (1024,4096)
  const float* Wa = (const float*)d_in[4];   // (1024,4096)
  const float* b  = (const float*)d_in[5];   // (4096,)
  float* out = (float*)d_out;                // (256,128,1024)

  char* ws = (char*)d_ws;
  bf16*  wT   = (bf16*)ws;   ws += (size_t)NOUT * KTOT * sizeof(bf16);   // 24 MB
  bf16*  act  = (bf16*)ws;   ws += (size_t)NB * KTOT * sizeof(bf16);     // 1.5 MB
  float* abuf = (float*)ws;  ws += (size_t)NB * NOUT * sizeof(float);    // 4 MB
  float* hbuf = (float*)ws;  ws += (size_t)NB * HH * sizeof(float);      // 1 MB
  float* cbuf = (float*)ws;                                              // 1 MB

  prep_weights_kernel<<<(NOUT * KTOT) / 256, 256, 0, stream>>>(Wx, Wh, Wa, wT);
  init_hc_kernel<<<(NB * HH) / 256, 256, 0, stream>>>(A, hbuf, cbuf);

  for (int t = 0; t < TT; ++t) {
    attn_pack_kernel<<<NB, 256, 0, stream>>>(x, A, hbuf, act, t);
    gemm_kernel<<<dim3(NOUT / 64, NB / 64), 256, 0, stream>>>(act, wT, abuf);
    gates_kernel<<<(NB * HH) / 256, 256, 0, stream>>>(abuf, b, hbuf, cbuf, out, t);
  }
}
